// MultiTaskLoss_11854109737211
// MI455X (gfx1250) — matrix operand order repaired and checked by reading, not compiled, not run on hardware
//
#include <hip/hip_runtime.h>
#include <hip/hip_bf16.h>
#include <stdint.h>

typedef __attribute__((ext_vector_type(16))) _Float16 v16h;
typedef __attribute__((ext_vector_type(8)))  float    v8f;

#define BLK     128
#define MAXV    10
#define NSTRIPE 64   // striped global accumulators: NSTRIPE sets of 4 floats in d_ws

__global__ void mtl_zero(float* acc) {
    const int i = threadIdx.x;
    if (i < NSTRIPE * 4) acc[i] = 0.0f;
}

__device__ __forceinline__ float shoelace4(const float2* p) {
    return 0.5f * ((p[0].x * p[1].y - p[1].x * p[0].y) +
                   (p[1].x * p[2].y - p[2].x * p[1].y) +
                   (p[2].x * p[3].y - p[3].x * p[2].y) +
                   (p[3].x * p[0].y - p[0].x * p[3].y));
}

__global__ __launch_bounds__(BLK) void mtl_main(
    const float*   __restrict__ logits,   // (B,N,C)
    const float*   __restrict__ pred,     // (B,N,8)
    const float*   __restrict__ gt,       // (B,N,8)
    const float*   __restrict__ pang,     // (B,N,1)
    const float*   __restrict__ gang,     // (B,N,1)
    const int*     __restrict__ gcls,     // (B,N)
    const uint8_t* __restrict__ vmask,    // (B,N) bool (1 byte)
    const float*   __restrict__ imgsz,    // (B,2)
    float*         __restrict__ acc,      // NSTRIPE*4 accumulators in d_ws
    int total, int N, int C)
{
    __shared__ float2 sA[BLK * MAXV];
    __shared__ float2 sB[BLK * MAXV];
    __shared__ float  sAcc[4];

    const int tid = threadIdx.x;
    if (tid < 4) sAcc[tid] = 0.0f;
    __syncthreads();

    const int idx = blockIdx.x * BLK + tid;
    float v_nll = 0.0f, v_obb = 0.0f, v_ang = 0.0f, v_m = 0.0f;

    if (idx < total) {
        const float m = vmask[idx] ? 1.0f : 0.0f;
        v_m = m;

        // ---- class loss: -log_softmax(logits)[gt_class] ----
        const float* lg = logits + (size_t)idx * (size_t)C;
        float mx = lg[0];
        for (int c = 1; c < C; ++c) mx = fmaxf(mx, lg[c]);
        float se = 0.0f;
        for (int c = 0; c < C; ++c) se += __expf(lg[c] - mx);
        const int cls = gcls[idx];
        v_nll = -(lg[cls] - mx - __logf(se)) * m;

        // ---- angle loss: smooth-L1 on cos(pred)-cos(gt) ----
        const float d  = __cosf(pang[idx]) - __cosf(gang[idx]);
        const float ad = fabsf(d);
        v_ang = (ad < 1.0f ? 0.5f * d * d : ad - 0.5f) * m;

        // ---- OBB loss: 1 - quad IoU via Sutherland-Hodgman clipping ----
        const int   b  = idx / N;
        const float sx = imgsz[2 * b + 0];
        const float sy = imgsz[2 * b + 1];
        const float* pp = pred + (size_t)idx * 8;
        const float* gg = gt   + (size_t)idx * 8;

        float2 P[4], Q[4];
        #pragma unroll
        for (int k = 0; k < 4; ++k) {
            P[k].x = pp[2 * k] * sx;  P[k].y = pp[2 * k + 1] * sy;
            Q[k].x = gg[2 * k] * sx;  Q[k].y = gg[2 * k + 1] * sy;
        }
        float aA = shoelace4(P);
        float aB = shoelace4(Q);
        if (aA < 0.0f) { float2 t0 = P[0]; P[0] = P[3]; P[3] = t0;
                         float2 t1 = P[1]; P[1] = P[2]; P[2] = t1; aA = -aA; }
        if (aB < 0.0f) { float2 t0 = Q[0]; Q[0] = Q[3]; Q[3] = t0;
                         float2 t1 = Q[1]; Q[1] = Q[2]; Q[2] = t1; aB = -aB; }

        // lane-major LDS polygon buffers (8B/lane * 32 lanes = 64 banks, conflict-free)
        float2* cur = sA + tid;
        float2* nxt = sB + tid;
        int n;

        // ---- clip pass 0: register quad P vs edge Q0->Q1 (static indexing) ----
        {
            const float2 e0 = Q[0], e1 = Q[1];
            const float ex = e1.x - e0.x, ey = e1.y - e0.y;
            float sv[4];
            #pragma unroll
            for (int k = 0; k < 4; ++k)
                sv[k] = ex * (P[k].y - e0.y) - ey * (P[k].x - e0.x);
            int mc = 0;
            #pragma unroll
            for (int j = 0; j < 4; ++j) {
                const int jp = (j + 3) & 3;
                const float2 pv = P[jp], pc = P[j];
                const float  sp = sv[jp], sc = sv[j];
                const bool ic = (sc >= 0.0f), ipv = (sp >= 0.0f);
                if (ic != ipv) {
                    const float t = sp / (sp - sc);
                    float2 ip;
                    ip.x = pv.x + t * (pc.x - pv.x);
                    ip.y = pv.y + t * (pc.y - pv.y);
                    if (mc < MAXV) cur[mc * BLK] = ip; ++mc;
                }
                if (ic) { if (mc < MAXV) cur[mc * BLK] = pc; ++mc; }
            }
            n = (mc > MAXV) ? MAXV : mc;
        }

        // ---- clip passes 1..3: LDS ping-pong, prev vertex tracked in regs ----
        for (int e = 1; e < 4 && n >= 3; ++e) {
            const float2 e0 = Q[e], e1 = Q[(e + 1) & 3];
            const float ex = e1.x - e0.x, ey = e1.y - e0.y;
            float2 pv = cur[(n - 1) * BLK];
            float  sp = ex * (pv.y - e0.y) - ey * (pv.x - e0.x);
            int mc = 0;
            for (int j = 0; j < n; ++j) {
                const float2 pc = cur[j * BLK];
                const float  sc = ex * (pc.y - e0.y) - ey * (pc.x - e0.x);
                const bool ic = (sc >= 0.0f), ipv = (sp >= 0.0f);
                if (ic != ipv) {
                    const float t = sp / (sp - sc);
                    float2 ip;
                    ip.x = pv.x + t * (pc.x - pv.x);
                    ip.y = pv.y + t * (pc.y - pv.y);
                    if (mc < MAXV) nxt[mc * BLK] = ip; ++mc;
                }
                if (ic) { if (mc < MAXV) nxt[mc * BLK] = pc; ++mc; }
                pv = pc; sp = sc;
            }
            float2* tmp = cur; cur = nxt; nxt = tmp;
            n = (mc > MAXV) ? MAXV : mc;
        }

        float inter = 0.0f;
        if (n >= 3) {
            const float2 first = cur[0];
            float2 pv = first;
            float s = 0.0f;
            for (int j = 1; j < n; ++j) {
                const float2 pc = cur[j * BLK];
                s += pv.x * pc.y - pc.x * pv.y;
                pv = pc;
            }
            s += pv.x * first.y - first.x * pv.y;
            inter = 0.5f * fabsf(s);
        }
        const float uni = aA + aB - inter;
        const float iou = (uni > 0.0f) ? inter / fmaxf(uni, 1e-9f) : 0.0f;
        v_obb = (1.0f - iou) * m;
    }

    // ---- wave32 reduction via WMMA tensor op ----
    // A: lane puts (qa,qb) in halves 0/1 -> K0/K1 (lanes 0-15) or K8/K9 (lanes 16-31).
    // B selector: col0 picks K0+K8, col1 picks K1+K9
    //   =>  D[m][0]=qa(m)+qa(m+16), D[m][1]=qb(m)+qb(m+16).
    const int lane = tid & 31;
    v16h a1 = {}; v16h a2 = {}; v16h bm = {};
    a1[0] = (_Float16)v_nll;  a1[1] = (_Float16)v_obb;
    a2[0] = (_Float16)v_ang;  a2[1] = (_Float16)v_m;
    const _Float16 one = (_Float16)1.0f;
    if ((lane & 15) == 0)      { bm[0] = one; bm[8] = one; }
    else if ((lane & 15) == 1) { bm[1] = one; bm[9] = one; }

    v8f zero8 = {};
    v8f c1 = __builtin_amdgcn_wmma_f32_16x16x32_f16(false, a1, false, bm,
                                                    (short)0, zero8, false, false);
    v8f c2 = __builtin_amdgcn_wmma_f32_16x16x32_f16(false, a2, false, bm,
                                                    (short)0, zero8, false, false);
    float s1 = 0.0f, s2 = 0.0f;
    #pragma unroll
    for (int k = 0; k < 8; ++k) { s1 += c1[k]; s2 += c2[k]; }
    s1 += __shfl_xor(s1, 16, 32);
    s2 += __shfl_xor(s2, 16, 32);

    if (lane == 0) { atomicAdd(&sAcc[0], s1); atomicAdd(&sAcc[2], s2); }
    if (lane == 1) { atomicAdd(&sAcc[1], s1); atomicAdd(&sAcc[3], s2); }
    __syncthreads();
    if (tid == 0) {
        float* slot = acc + (size_t)((blockIdx.x & (NSTRIPE - 1)) * 4);
        atomicAdd(&slot[0], sAcc[0]);
        atomicAdd(&slot[1], sAcc[1]);
        atomicAdd(&slot[2], sAcc[2]);
        atomicAdd(&slot[3], sAcc[3]);
    }
}

__global__ void mtl_finalize(const float* __restrict__ acc, float* __restrict__ out) {
    __shared__ float tmp[4];
    const int k = threadIdx.x;
    if (k < 4) {
        float s = 0.0f;
        for (int j = 0; j < NSTRIPE; ++j) s += acc[j * 4 + k];
        tmp[k] = s;
    }
    __syncthreads();
    if (k == 0) {
        const float cnt = fmaxf(tmp[3], 1.0f);
        const float lc = tmp[0] / cnt;
        const float lo = tmp[1] / cnt;
        const float la = tmp[2] / cnt;
        out[0] = lc + lo + la;
        out[1] = lc;
        out[2] = lo;
        out[3] = la;
    }
}

extern "C" void kernel_launch(void* const* d_in, const int* in_sizes, int n_in,
                              void* d_out, int out_size, void* d_ws, size_t ws_size,
                              hipStream_t stream) {
    const float*   logits = (const float*)  d_in[0];
    const float*   pred   = (const float*)  d_in[1];
    const float*   gtb    = (const float*)  d_in[2];
    const float*   pang   = (const float*)  d_in[3];
    const float*   gang   = (const float*)  d_in[4];
    const int*     gcls   = (const int*)    d_in[5];
    const uint8_t* vmask  = (const uint8_t*)d_in[6];
    const float*   imgsz  = (const float*)  d_in[7];

    const int total = in_sizes[5];            // B*N
    const int Bsz   = in_sizes[7] / 2;        // image_size is (B,2)
    const int N     = total / Bsz;
    const int C     = in_sizes[0] / total;

    float* acc = (float*)d_ws;                // NSTRIPE*4 f32 accumulators
    float* out = (float*)d_out;

    mtl_zero<<<1, NSTRIPE * 4, 0, stream>>>(acc);
    const int grid = (total + BLK - 1) / BLK;
    mtl_main<<<grid, BLK, 0, stream>>>(logits, pred, gtb, pang, gang, gcls,
                                       vmask, imgsz, acc, total, N, C);
    mtl_finalize<<<1, 32, 0, stream>>>(acc, out);
}
